// ConformerEncoder_41807211659776
// MI455X (gfx1250) — matrix operand order repaired and checked by reading, not compiled, not run on hardware
//
#include <hip/hip_runtime.h>

// ---------------------------------------------------------------------------
// Types & helpers
// ---------------------------------------------------------------------------
typedef __attribute__((ext_vector_type(16))) __bf16    v16bf;
typedef __attribute__((ext_vector_type(2)))  __bf16    v2bf;
typedef __attribute__((ext_vector_type(8)))  float     v8f;
typedef __attribute__((ext_vector_type(4)))  float     v4flt;
typedef __attribute__((ext_vector_type(4)))  unsigned  v4u;
typedef __attribute__((ext_vector_type(2)))  unsigned  v2u;

union FragU { v16bf v; unsigned u[8]; v4u q[2]; unsigned short s[16]; };

__device__ __forceinline__ unsigned short f2bf(float f) {
  unsigned u = __builtin_bit_cast(unsigned, f);
  u += 0x7FFFu + ((u >> 16) & 1u);            // round-to-nearest-even
  return (unsigned short)(u >> 16);
}
// pack two fp32 -> one dword of bf16x2 (prefer HW v_cvt_pk_bf16_f32)
__device__ __forceinline__ unsigned pk2bf(float lo, float hi) {
#if __has_builtin(__builtin_amdgcn_cvt_pk_bf16_f32)
  v2bf r = __builtin_amdgcn_cvt_pk_bf16_f32(lo, hi);
  return __builtin_bit_cast(unsigned, r);
#else
  return (unsigned)f2bf(lo) | ((unsigned)f2bf(hi) << 16);
#endif
}
__device__ __forceinline__ v8f v8f_zero() {
  v8f z = {0.f, 0.f, 0.f, 0.f, 0.f, 0.f, 0.f, 0.f};
  return z;
}
__device__ __forceinline__ v8f wmma_bf16(v16bf a, v16bf b, v8f c) {
  return __builtin_amdgcn_wmma_f32_16x16x32_bf16(false, a, false, b,
                                                 (short)0, c, false, false);
}

#define APITCH 72   // ushorts per LDS row (144B: 16B-aligned, conflict-free)

// A-fragment (16x32 bf16): per lane two contiguous 16B runs.
__device__ __forceinline__ v16bf load_a_frag(const unsigned short* As,
                                             int mBase, int kk, int lane) {
  FragU f;
  int m = mBase + (lane & 15), hi = lane >> 4;
  f.q[0] = *(const v4u*)(As + m * APITCH + kk + 8 * hi);
  f.q[1] = *(const v4u*)(As + m * APITCH + kk + 16 + 8 * hi);
  return f.v;
}
// B-fragment (32x16 bf16) from LDS stored TRANSPOSED [n][k].
__device__ __forceinline__ v16bf load_b_frag(const unsigned short* Bt,
                                             int nBase, int kk, int lane) {
  FragU f;
  int n = nBase + (lane & 15), kh = lane >> 4;
  f.q[0] = *(const v4u*)(Bt + n * APITCH + kk + 8 * kh);
  f.q[1] = *(const v4u*)(Bt + n * APITCH + kk + 16 + 8 * kh);
  return f.v;
}

__device__ __forceinline__ void epilogue_store(v8f acc, int mTop, int nTop,
                                               const float* bias,
                                               const float* resid, float* C,
                                               int N, int act, float rs,
                                               int lane) {
  int n = nTop + (lane & 15);
  float bv = bias ? bias[n] : 0.f;
#pragma unroll
  for (int r = 0; r < 8; ++r) {
    int m = mTop + r + 8 * (lane >> 4);
    float v = acc[r] + bv;
    if (act == 1) v = fmaxf(v, 0.f);
    else if (act == 2) v = v / (1.f + __expf(-v));    // SiLU
    float o = rs * v;
    if (resid) o += resid[(size_t)m * N + n];
    C[(size_t)m * N + n] = o;
  }
}

// ---------------------------------------------------------------------------
// Generic GEMM: C[M,N] = epilogue(A[M,K] fp32  x  Bwt[N,K] bf16 + bias)
// Weights pre-transposed to [N][K]. block = 8 waves; BM=128,BN=64,BK=64.
// Requires M%128==0, N%64==0, K%64==0.
// ---------------------------------------------------------------------------
__launch_bounds__(256)
__global__ void gemm_bf16w(const float* __restrict__ A,
                           const unsigned short* __restrict__ Bwt,
                           const float* __restrict__ bias,
                           const float* __restrict__ resid,
                           float* __restrict__ C,
                           int M, int N, int K, int act, float rs) {
  __shared__ unsigned short As[128 * APITCH];
  __shared__ unsigned short Bt[64 * APITCH];
  const int tid = threadIdx.x;
  const int lane = tid & 31;
  const int wave = tid >> 5;
  const int wm = wave >> 1, wn = wave & 1;
  const int m0 = blockIdx.y * 128;
  const int n0 = blockIdx.x * 64;

  v8f a00 = v8f_zero(), a01 = v8f_zero(), a10 = v8f_zero(), a11 = v8f_zero();

  for (int k0 = 0; k0 < K; k0 += 64) {
    __syncthreads();
    // stage A: 128x64 fp32 -> bf16, 8 x float4 per thread
#pragma unroll
    for (int i = 0; i < 8; ++i) {
      int idx = tid + 256 * i;          // 2048 chunks of 4 floats
      int r = idx >> 4;                 // 16 chunks per row
      int c4 = (idx & 15) * 4;
      v4flt f = *(const v4flt*)(A + (size_t)(m0 + r) * K + k0 + c4);
      v2u pk;
      pk[0] = pk2bf(f[0], f[1]);
      pk[1] = pk2bf(f[2], f[3]);
      *(v2u*)(As + r * APITCH + c4) = pk;
    }
    // stage B (already [n][k] in global): 2 x b128 per thread
#pragma unroll
    for (int i = 0; i < 2; ++i) {
      int idx = tid + 256 * i;          // 512 chunks of 8 ushorts
      int n = idx >> 3;                 // 8 chunks per row
      int kc8 = (idx & 7) * 8;
      *(v4u*)(Bt + n * APITCH + kc8) =
          *(const v4u*)(Bwt + (size_t)(n0 + n) * K + k0 + kc8);
    }
    if (k0 + 64 < K)                    // emits global_prefetch_b8
      __builtin_prefetch(Bwt + (size_t)(n0 + (tid & 63)) * K + k0 + 64, 0, 0);
    __syncthreads();

#pragma unroll
    for (int kk = 0; kk < 64; kk += 32) {
      v16bf aF0 = load_a_frag(As, wm * 32,      kk, lane);
      v16bf aF1 = load_a_frag(As, wm * 32 + 16, kk, lane);
      v16bf bF0 = load_b_frag(Bt, wn * 32,      kk, lane);
      v16bf bF1 = load_b_frag(Bt, wn * 32 + 16, kk, lane);
      a00 = wmma_bf16(aF0, bF0, a00);
      a01 = wmma_bf16(aF0, bF1, a01);
      a10 = wmma_bf16(aF1, bF0, a10);
      a11 = wmma_bf16(aF1, bF1, a11);
    }
  }

  int mT = m0 + wm * 32, nT = n0 + wn * 32;
  epilogue_store(a00, mT,      nT,      bias, resid, C, N, act, rs, lane);
  epilogue_store(a01, mT,      nT + 16, bias, resid, C, N, act, rs, lane);
  epilogue_store(a10, mT + 16, nT,      bias, resid, C, N, act, rs, lane);
  epilogue_store(a11, mT + 16, nT + 16, bias, resid, C, N, act, rs, lane);
}

// ---------------------------------------------------------------------------
// conv2 implicit GEMM: M=8192 rows=(b,t2,f2), K=6912=ic*9+kh*3+kw, N=768.
// A gathered (im2col on the fly) from NHWC bf16 Y1[b][t1(512)][f1(32)][ic].
// Weights: sub_w2 is [oc][k] already == [N][K] transposed layout.
// ---------------------------------------------------------------------------
__launch_bounds__(256)
__global__ void gemm_conv2(const unsigned short* __restrict__ Y1,
                           const unsigned short* __restrict__ Bwt,
                           const float* __restrict__ bias,
                           float* __restrict__ C) {
  const int N = 768, K = 6912;
  __shared__ unsigned short As[128 * APITCH];
  __shared__ unsigned short Bt[64 * APITCH];
  const int tid = threadIdx.x;
  const int lane = tid & 31;
  const int wave = tid >> 5;
  const int wm = wave >> 1, wn = wave & 1;
  const int m0 = blockIdx.y * 128;
  const int n0 = blockIdx.x * 64;

  v8f a00 = v8f_zero(), a01 = v8f_zero(), a10 = v8f_zero(), a11 = v8f_zero();

  for (int k0 = 0; k0 < K; k0 += 64) {
    __syncthreads();
#pragma unroll
    for (int i = 0; i < 32; ++i) {      // im2col gather, 8192 elements
      int idx = tid + 256 * i;
      int r = idx >> 6, c = idx & 63;
      int m = m0 + r;
      int k = k0 + c;
      int b  = m >> 12;
      int rem = m & 4095;
      int t2 = rem >> 4, f2 = rem & 15;
      int ic = k / 9;
      int r9 = k - ic * 9;
      int kh = r9 / 3, kw = r9 - kh * 3;
      int t1 = 2 * t2 + kh - 1;
      int f1 = 2 * f2 + kw - 1;
      unsigned short v = 0;
      if ((unsigned)t1 < 512u && (unsigned)f1 < 32u)
        v = Y1[(((size_t)b * 512 + t1) * 32 + f1) * 768 + ic];
      As[r * APITCH + c] = v;
    }
#pragma unroll
    for (int i = 0; i < 2; ++i) {
      int idx = tid + 256 * i;
      int n = idx >> 3;
      int kc8 = (idx & 7) * 8;
      *(v4u*)(Bt + n * APITCH + kc8) =
          *(const v4u*)(Bwt + (size_t)(n0 + n) * K + k0 + kc8);
    }
    __syncthreads();
#pragma unroll
    for (int kk = 0; kk < 64; kk += 32) {
      v16bf aF0 = load_a_frag(As, wm * 32,      kk, lane);
      v16bf aF1 = load_a_frag(As, wm * 32 + 16, kk, lane);
      v16bf bF0 = load_b_frag(Bt, wn * 32,      kk, lane);
      v16bf bF1 = load_b_frag(Bt, wn * 32 + 16, kk, lane);
      a00 = wmma_bf16(aF0, bF0, a00);
      a01 = wmma_bf16(aF0, bF1, a01);
      a10 = wmma_bf16(aF1, bF0, a10);
      a11 = wmma_bf16(aF1, bF1, a11);
    }
  }
  int mT = m0 + wm * 32, nT = n0 + wn * 32;
  epilogue_store(a00, mT,      nT,      bias, nullptr, C, N, 1, 1.f, lane);
  epilogue_store(a01, mT,      nT + 16, bias, nullptr, C, N, 1, 1.f, lane);
  epilogue_store(a10, mT + 16, nT,      bias, nullptr, C, N, 1, 1.f, lane);
  epilogue_store(a11, mT + 16, nT + 16, bias, nullptr, C, N, 1, 1.f, lane);
}

// ---------------------------------------------------------------------------
// Attention: one workgroup per (b,h). T=256, DK=48 padded to 64.
// LDS: Kb[256][64] bf16 | VbT[64][256] bf16 | per-wave S 16x256 f32.
// ---------------------------------------------------------------------------
__launch_bounds__(256)
__global__ void attention_kernel(const unsigned short* __restrict__ Qr,
                                 const unsigned short* __restrict__ Kr,
                                 const unsigned short* __restrict__ Vr,
                                 float* __restrict__ O) {
  extern __shared__ char smem[];
  unsigned short* Kb  = (unsigned short*)smem;              // [t][d]
  unsigned short* VbT = (unsigned short*)(smem + 32768);    // [d][t]
  float* Sall = (float*)(smem + 65536);

  const int bh = blockIdx.x;
  const int b = bh >> 4, h = bh & 15;
  const int tid = threadIdx.x, lane = tid & 31, wave = tid >> 5;
  const int mlane = lane & 15, hi = lane >> 4;
  const size_t base = (size_t)bh * 256 * 64;

#pragma unroll
  for (int j = 0; j < 8; ++j) {         // 2048 chunks of 8 ushorts
    int i = (tid + 256 * j) * 8;
    *(v4u*)(Kb + i) = *(const v4u*)(Kr + base + i);
    union { v4u q; unsigned short s[8]; } uv;
    uv.q = *(const v4u*)(Vr + base + i);
    int t = i >> 6, d0 = i & 63;
#pragma unroll
    for (int e = 0; e < 8; ++e) VbT[(d0 + e) * 256 + t] = uv.s[e];
  }
  __syncthreads();

  float* S = Sall + wave * (16 * 256);
  unsigned short* Ps = (unsigned short*)S;
  const float scale = 0.144337567297406f;  // 1/sqrt(48)

  for (int qi = 0; qi < 2; ++qi) {
    int qt = wave * 2 + qi;
    // Q fragments straight from global (row-major bf16, padded to 64)
    FragU aQ[2];
#pragma unroll
    for (int kc = 0; kc < 2; ++kc) {
      const unsigned short* qp =
          Qr + base + (size_t)(qt * 16 + mlane) * 64 + kc * 32 + 8 * hi;
      aQ[kc].q[0] = *(const v4u*)qp;
      aQ[kc].q[1] = *(const v4u*)(qp + 16);
    }
    // scores S = Q K^T * scale (16 x 256)
    for (int nt = 0; nt < 16; ++nt) {
      v8f acc = v8f_zero();
#pragma unroll
      for (int kc = 0; kc < 2; ++kc) {
        FragU bK;
        const unsigned short* kp =
            Kb + (nt * 16 + mlane) * 64 + kc * 32 + 8 * hi;
        bK.q[0] = *(const v4u*)kp;
        bK.q[1] = *(const v4u*)(kp + 16);
        acc = wmma_bf16(aQ[kc].v, bK.v, acc);
      }
#pragma unroll
      for (int r = 0; r < 8; ++r)
        S[(r + 8 * hi) * 256 + nt * 16 + mlane] = acc[r] * scale;
    }
    __syncthreads();

    // softmax per row; rewrite P as packed bf16 (aliasing-safe ordering)
    for (int r = 0; r < 16; ++r) {
      float mx = -1e30f;
#pragma unroll
      for (int i = 0; i < 8; ++i) mx = fmaxf(mx, S[r * 256 + lane + 32 * i]);
#pragma unroll
      for (int off = 16; off; off >>= 1) mx = fmaxf(mx, __shfl_xor(mx, off));
      float sum = 0.f;
#pragma unroll
      for (int i = 0; i < 8; ++i) {
        float e = __expf(S[r * 256 + lane + 32 * i] - mx);
        S[r * 256 + lane + 32 * i] = e;
        sum += e;
      }
#pragma unroll
      for (int off = 16; off; off >>= 1) sum += __shfl_xor(sum, off);
      float inv = 1.f / sum;
#pragma unroll
      for (int i = 0; i < 8; ++i) {
        int c = lane + 32 * i;
        Ps[r * 256 + c] = f2bf(S[r * 256 + c] * inv);
      }
    }
    __syncthreads();

    // O = P (16x256) @ V (256x64); only d<48 stored
    v8f o0 = v8f_zero(), o1 = v8f_zero(), o2 = v8f_zero();
    for (int kc = 0; kc < 8; ++kc) {
      FragU aP;
      const unsigned short* pp = Ps + mlane * 256 + kc * 32 + 8 * hi;
      aP.q[0] = *(const v4u*)pp;
      aP.q[1] = *(const v4u*)(pp + 16);
#pragma unroll
      for (int nt = 0; nt < 3; ++nt) {
        FragU bV;
        const unsigned short* vp =
            VbT + (nt * 16 + mlane) * 256 + kc * 32 + 8 * hi;
        bV.q[0] = *(const v4u*)vp;
        bV.q[1] = *(const v4u*)(vp + 16);
        v8f acc = (nt == 0) ? o0 : (nt == 1) ? o1 : o2;
        acc = wmma_bf16(aP.v, bV.v, acc);
        if (nt == 0) o0 = acc; else if (nt == 1) o1 = acc; else o2 = acc;
      }
    }
#pragma unroll
    for (int nt = 0; nt < 3; ++nt) {
      v8f acc = (nt == 0) ? o0 : (nt == 1) ? o1 : o2;
      int d = nt * 16 + mlane;
#pragma unroll
      for (int r = 0; r < 8; ++r) {
        int trow = qt * 16 + r + 8 * hi;
        O[((size_t)(b * 256 + trow)) * 768 + h * 48 + d] = acc[r];
      }
    }
    __syncthreads();
  }
}

// ---------------------------------------------------------------------------
// Small support kernels
// ---------------------------------------------------------------------------
__global__ void calc_outlen(const int* __restrict__ lengths, int* ws_len,
                            int* out_tail) {
  int b = threadIdx.x;
  if (b < 2) {
    int l = lengths[b];
    l = (l + 1) >> 1;   // floor((l-1)/2)+1
    l = (l + 1) >> 1;
    ws_len[b] = l;
    out_tail[b] = l;
  }
}

// conv1: image(B,1,1024,64) from features(B,64,1024); 3x3 s2 p1, 768ch, ReLU
// output NHWC bf16: Y1[b][t1(512)][f1(32)][oc(768)]
__global__ void conv1_relu(const float* __restrict__ F,
                           const float* __restrict__ W,
                           const float* __restrict__ bias,
                           unsigned short* __restrict__ Y1) {
  int idx = blockIdx.x * 256 + threadIdx.x;
  if (idx >= 2 * 512 * 32 * 768) return;
  int oc = idx % 768;
  int t = idx / 768;
  int f1 = t % 32; t /= 32;
  int t1 = t % 512;
  int b = t / 512;
  float s = bias[oc];
#pragma unroll
  for (int kh = 0; kh < 3; ++kh) {
    int hh = 2 * t1 + kh - 1;
    if ((unsigned)hh >= 1024u) continue;
#pragma unroll
    for (int kw = 0; kw < 3; ++kw) {
      int wwi = 2 * f1 + kw - 1;
      if ((unsigned)wwi >= 64u) continue;
      s += W[oc * 9 + kh * 3 + kw] * F[((size_t)b * 64 + wwi) * 1024 + hh];
    }
  }
  Y1[idx] = f2bf(fmaxf(s, 0.f));
}

// straight fp32 -> bf16 convert (layout preserved)
__global__ void cvt_bf16_arr(const float* __restrict__ src,
                             unsigned short* __restrict__ dst, int n) {
  int i = blockIdx.x * 256 + threadIdx.x;
  if (i < n) dst[i] = f2bf(src[i]);
}

// tiled transpose+convert: src[K][N] fp32 -> dst[N][K] bf16. K,N % 32 == 0.
__global__ void cvt_tr(const float* __restrict__ src,
                       unsigned short* __restrict__ dst, int K, int N) {
  __shared__ unsigned short tile[32][33];
  int tx = threadIdx.x & 31, ty = threadIdx.x >> 5;
  int nIn = blockIdx.x * 32 + tx;
#pragma unroll
  for (int j = 0; j < 32; j += 8) {
    int k = blockIdx.y * 32 + ty + j;
    tile[ty + j][tx] = f2bf(src[(size_t)k * N + nIn]);
  }
  __syncthreads();
  int kOut = blockIdx.y * 32 + tx;
#pragma unroll
  for (int j = 0; j < 32; j += 8) {
    int n = blockIdx.x * 32 + ty + j;
    dst[(size_t)n * K + kOut] = tile[tx][ty + j];
  }
}

// c2out[(b,t2,f2)][oc] -> x0[(b,t2)][oc*16+f2]
__global__ void repack_pre(const float* __restrict__ c2,
                           float* __restrict__ x0) {
  int i = blockIdx.x * 256 + threadIdx.x;
  if (i >= 512 * 12288) return;
  int row = i / 12288, col = i % 12288;
  int oc = col >> 4, f2 = col & 15;
  x0[i] = c2[(((size_t)row * 16) + f2) * 768 + oc];
}

__global__ void layernorm_k(const float* __restrict__ x,
                            const float* __restrict__ g,
                            const float* __restrict__ bta,
                            float* __restrict__ y) {
  __shared__ float red[256];
  int row = blockIdx.x, tid = threadIdx.x;
  const float* xr = x + (size_t)row * 768;
  float v0 = xr[tid], v1 = xr[tid + 256], v2 = xr[tid + 512];
  red[tid] = v0 + v1 + v2;
  __syncthreads();
  for (int o = 128; o; o >>= 1) {
    if (tid < o) red[tid] += red[tid + o];
    __syncthreads();
  }
  float mean = red[0] * (1.f / 768.f);
  __syncthreads();
  float d0 = v0 - mean, d1 = v1 - mean, d2 = v2 - mean;
  red[tid] = d0 * d0 + d1 * d1 + d2 * d2;
  __syncthreads();
  for (int o = 128; o; o >>= 1) {
    if (tid < o) red[tid] += red[tid + o];
    __syncthreads();
  }
  float inv = rsqrtf(red[0] * (1.f / 768.f) + 1e-5f);
  float* yr = y + (size_t)row * 768;
  yr[tid]       = d0 * inv * g[tid]       + bta[tid];
  yr[tid + 256] = d1 * inv * g[tid + 256] + bta[tid + 256];
  yr[tid + 512] = d2 * inv * g[tid + 512] + bta[tid + 512];
}

// rotary + pack q,k,v into [bh][t][64] bf16 (DK 48 zero-padded)
__global__ void rotary_pack(const float* __restrict__ q,
                            const float* __restrict__ k,
                            const float* __restrict__ v,
                            unsigned short* __restrict__ Qr,
                            unsigned short* __restrict__ Kr,
                            unsigned short* __restrict__ Vr) {
  int idx = blockIdx.x * 256 + threadIdx.x;
  if (idx >= 2 * 16 * 256 * 64) return;
  int d = idx & 63;
  int t = (idx >> 6) & 255;
  int h = (idx >> 14) & 15;
  int b = idx >> 18;
  if (d >= 48) { Qr[idx] = 0; Kr[idx] = 0; Vr[idx] = 0; return; }
  size_t src = ((size_t)(b * 256 + t)) * 768 + h * 48 + d;
  int i = (d < 24) ? d : d - 24;
  float ang = (float)t * __powf(5000.f, -(float)(2 * i) / 48.f);
  float c = __cosf(ang), s = __sinf(ang);
  float qv = q[src], kv = k[src];
  float qr = (d < 24) ? -q[src + 24] : q[src - 24];
  float kr = (d < 24) ? -k[src + 24] : k[src - 24];
  Qr[idx] = f2bf(qv * c + qr * s);
  Kr[idx] = f2bf(kv * c + kr * s);
  Vr[idx] = f2bf(v[src]);
}

__global__ void glu_mask(const float* __restrict__ hIn, const int* outlen,
                         float* __restrict__ out) {
  int idx = blockIdx.x * 256 + threadIdx.x;
  if (idx >= 512 * 768) return;
  int m = idx / 768, c = idx - m * 768;
  int b = m >> 8, t = m & 255;
  float a = hIn[(size_t)m * 1536 + c];
  float g = hIn[(size_t)m * 1536 + 768 + c];
  float val = a / (1.f + __expf(-g));
  out[idx] = (t >= outlen[b]) ? 0.f : val;
}

__global__ void dwconv_k(const float* __restrict__ g,
                         const float* __restrict__ w,
                         const float* __restrict__ bias,
                         float* __restrict__ out) {
  int idx = blockIdx.x * 256 + threadIdx.x;
  if (idx >= 512 * 768) return;
  int m = idx / 768, c = idx - m * 768;
  int b = m >> 8, t = m & 255;
  float s = bias[c];
#pragma unroll
  for (int j = 0; j < 31; ++j) {
    int tt = t + j - 15;
    if ((unsigned)tt >= 256u) continue;
    s += g[((size_t)(b * 256 + tt)) * 768 + c] * w[c * 31 + j];
  }
  out[idx] = s;
}

__global__ void bn_stats(const float* __restrict__ x,
                         float* __restrict__ st) {
  int c = blockIdx.x * 256 + threadIdx.x;
  if (c >= 768) return;
  float s = 0.f, s2 = 0.f;
  for (int m = 0; m < 512; ++m) {
    float v = x[(size_t)m * 768 + c];
    s += v;
    s2 += v * v;
  }
  float mean = s * (1.f / 512.f);
  float var = s2 * (1.f / 512.f) - mean * mean;
  st[c] = mean;
  st[768 + c] = rsqrtf(var + 1e-5f);
}

__global__ void bn_apply_silu(const float* __restrict__ x,
                              const float* __restrict__ st,
                              const float* __restrict__ g,
                              const float* __restrict__ b,
                              float* __restrict__ y) {
  int idx = blockIdx.x * 256 + threadIdx.x;
  if (idx >= 512 * 768) return;
  int c = idx % 768;
  float v = (x[idx] - st[c]) * st[768 + c] * g[c] + b[c];
  y[idx] = v / (1.f + __expf(-v));
}

__global__ void transpose_out(const float* __restrict__ x,
                              float* __restrict__ out) {
  int idx = blockIdx.x * 256 + threadIdx.x;
  if (idx >= 2 * 768 * 256) return;
  int t = idx & 255;
  int c = (idx >> 8) % 768;
  int b = idx / (768 * 256);
  out[idx] = x[((size_t)(b * 256 + t)) * 768 + c];
}

// ---------------------------------------------------------------------------
// Host orchestration
// ---------------------------------------------------------------------------
extern "C" void kernel_launch(void* const* d_in, const int* in_sizes, int n_in,
                              void* d_out, int out_size, void* d_ws,
                              size_t ws_size, hipStream_t stream) {
  (void)in_sizes; (void)n_in; (void)out_size; (void)ws_size;
  const float* F        = (const float*)d_in[0];
  const int*   lengths  = (const int*)d_in[1];
  int pi = 2;
  const float* sub_w1   = (const float*)d_in[pi++];
  const float* sub_b1   = (const float*)d_in[pi++];
  const float* sub_w2   = (const float*)d_in[pi++];
  const float* sub_b2   = (const float*)d_in[pi++];
  const float* sub_ow   = (const float*)d_in[pi++];
  const float* sub_ob   = (const float*)d_in[pi++];
  const float* ln_ff1_g = (const float*)d_in[pi++];
  const float* ln_ff1_b = (const float*)d_in[pi++];
  const float* ln_att_g = (const float*)d_in[pi++];
  const float* ln_att_b = (const float*)d_in[pi++];
  const float* ln_cv_g  = (const float*)d_in[pi++];
  const float* ln_cv_b  = (const float*)d_in[pi++];
  const float* ln_ff2_g = (const float*)d_in[pi++];
  const float* ln_ff2_b = (const float*)d_in[pi++];
  const float* ln_out_g = (const float*)d_in[pi++];
  const float* ln_out_b = (const float*)d_in[pi++];
  const float* ff1_w1   = (const float*)d_in[pi++];
  const float* ff1_b1   = (const float*)d_in[pi++];
  const float* ff1_w2   = (const float*)d_in[pi++];
  const float* ff1_b2   = (const float*)d_in[pi++];
  const float* ff2_w1   = (const float*)d_in[pi++];
  const float* ff2_b1   = (const float*)d_in[pi++];
  const float* ff2_w2   = (const float*)d_in[pi++];
  const float* ff2_b2   = (const float*)d_in[pi++];
  const float* wq       = (const float*)d_in[pi++];
  const float* bq       = (const float*)d_in[pi++];
  const float* wk       = (const float*)d_in[pi++];
  const float* bk       = (const float*)d_in[pi++];
  const float* wv       = (const float*)d_in[pi++];
  const float* bv       = (const float*)d_in[pi++];
  const float* wo       = (const float*)d_in[pi++];
  const float* bo       = (const float*)d_in[pi++];
  const float* pw1_w    = (const float*)d_in[pi++];
  const float* pw1_b    = (const float*)d_in[pi++];
  const float* dw_w     = (const float*)d_in[pi++];
  const float* dw_b     = (const float*)d_in[pi++];
  const float* bn_g     = (const float*)d_in[pi++];
  const float* bn_b     = (const float*)d_in[pi++];
  const float* pw2_w    = (const float*)d_in[pi++];
  const float* pw2_b    = (const float*)d_in[pi++];

  // ------ workspace bump allocator ------
  char* wsb = (char*)d_ws;
  size_t off = 0;
  auto alloc = [&](size_t bytes) {
    void* p = wsb + off;
    off += (bytes + 255) & ~(size_t)255;
    return p;
  };
  int*            outlen = (int*)alloc(16);
  float*          xa  = (float*)alloc((size_t)512 * 768 * 4);
  float*          xb  = (float*)alloc((size_t)512 * 768 * 4);
  float*          h1  = (float*)alloc((size_t)512 * 3072 * 4);
  float*          h2  = (float*)alloc((size_t)512 * 768 * 4);
  float*          qb  = (float*)alloc((size_t)512 * 768 * 4);
  float*          kb  = (float*)alloc((size_t)512 * 768 * 4);
  float*          vb  = (float*)alloc((size_t)512 * 768 * 4);
  float*          ob  = (float*)alloc((size_t)512 * 768 * 4);
  unsigned short* Qr  = (unsigned short*)alloc((size_t)524288 * 2);
  unsigned short* Kr  = (unsigned short*)alloc((size_t)524288 * 2);
  unsigned short* Vr  = (unsigned short*)alloc((size_t)524288 * 2);
  float*          glu = (float*)alloc((size_t)512 * 768 * 4);
  float*          dwo = (float*)alloc((size_t)512 * 768 * 4);
  float*          bst = (float*)alloc((size_t)768 * 2 * 4);
  unsigned short* wbuf = (unsigned short*)alloc((size_t)12288 * 768 * 2);
  unsigned short* Y1  = (unsigned short*)alloc((size_t)25165824 * 2);
  float*          c2o = (float*)alloc((size_t)8192 * 768 * 4);
  float*          x0l = (float*)alloc((size_t)512 * 12288 * 4);

  auto gemm = [&](const float* A, const unsigned short* Bwt, const float* bias,
                  const float* resid, float* C, int M, int N, int K, int act,
                  float rs) {
    dim3 grid(N / 64, M / 128);
    gemm_bf16w<<<grid, 256, 0, stream>>>(A, Bwt, bias, resid, C, M, N, K, act,
                                         rs);
  };
  // convert + transpose weight [K][N] fp32 -> [N][K] bf16 into wbuf
  auto cvtw = [&](const float* src, int K, int N) {
    cvt_tr<<<dim3(N / 32, K / 32), 256, 0, stream>>>(src, wbuf, K, N);
  };

  // ------ pre-encoder ------
  calc_outlen<<<1, 32, 0, stream>>>(lengths, outlen, (int*)d_out + 393216);
  conv1_relu<<<(25165824 + 255) / 256, 256, 0, stream>>>(F, sub_w1, sub_b1, Y1);
  // sub_w2 is [oc][ic*9+kh*3+kw] == [N][K]: straight convert
  cvt_bf16_arr<<<(768 * 6912 + 255) / 256, 256, 0, stream>>>(sub_w2, wbuf,
                                                             768 * 6912);
  gemm_conv2<<<dim3(12, 64), 256, 0, stream>>>(Y1, wbuf, sub_b2, c2o);
  repack_pre<<<(512 * 12288 + 255) / 256, 256, 0, stream>>>(c2o, x0l);
  cvtw(sub_ow, 12288, 768);
  gemm(x0l, wbuf, sub_ob, nullptr, xa, 512, 768, 12288, 0, 1.f);

  // ------ conformer layers ------
  float* A = xa;
  float* B = xb;
  for (int i = 0; i < 8; ++i) {
    size_t oD = (size_t)i * 768;
    size_t oDF = (size_t)i * 3072;
    size_t oDD = (size_t)i * 768 * 768;
    // macaron FFN 1 (half residual)
    layernorm_k<<<512, 256, 0, stream>>>(A, ln_ff1_g + oD, ln_ff1_b + oD, h2);
    cvtw(ff1_w1 + (size_t)i * 768 * 3072, 768, 3072);
    gemm(h2, wbuf, ff1_b1 + oDF, nullptr, h1, 512, 3072, 768, 2, 1.f);
    cvtw(ff1_w2 + (size_t)i * 3072 * 768, 3072, 768);
    gemm(h1, wbuf, ff1_b2 + oD, A, B, 512, 768, 3072, 0, 0.5f);
    // rotary attention
    layernorm_k<<<512, 256, 0, stream>>>(B, ln_att_g + oD, ln_att_b + oD, h2);
    cvtw(wq + oDD, 768, 768);
    gemm(h2, wbuf, bq + oD, nullptr, qb, 512, 768, 768, 0, 1.f);
    cvtw(wk + oDD, 768, 768);
    gemm(h2, wbuf, bk + oD, nullptr, kb, 512, 768, 768, 0, 1.f);
    cvtw(wv + oDD, 768, 768);
    gemm(h2, wbuf, bv + oD, nullptr, vb, 512, 768, 768, 0, 1.f);
    rotary_pack<<<(524288 + 255) / 256, 256, 0, stream>>>(qb, kb, vb, Qr, Kr,
                                                          Vr);
    attention_kernel<<<32, 256, 196608, stream>>>(Qr, Kr, Vr, ob);
    cvtw(wo + oDD, 768, 768);
    gemm(ob, wbuf, bo + oD, B, A, 512, 768, 768, 0, 1.f);
    // conv module
    layernorm_k<<<512, 256, 0, stream>>>(A, ln_cv_g + oD, ln_cv_b + oD, h2);
    cvtw(pw1_w + (size_t)i * 768 * 1536, 768, 1536);
    gemm(h2, wbuf, pw1_b + (size_t)i * 1536, nullptr, h1, 512, 1536, 768, 0,
         1.f);
    glu_mask<<<1536, 256, 0, stream>>>(h1, outlen, glu);
    dwconv_k<<<1536, 256, 0, stream>>>(glu, dw_w + (size_t)i * 768 * 31,
                                       dw_b + oD, dwo);
    bn_stats<<<3, 256, 0, stream>>>(dwo, bst);
    bn_apply_silu<<<1536, 256, 0, stream>>>(dwo, bst, bn_g + oD, bn_b + oD, h2);
    cvtw(pw2_w + oDD, 768, 768);
    gemm(h2, wbuf, pw2_b + oD, A, B, 512, 768, 768, 0, 1.f);
    // macaron FFN 2 (half residual)
    layernorm_k<<<512, 256, 0, stream>>>(B, ln_ff2_g + oD, ln_ff2_b + oD, h2);
    cvtw(ff2_w1 + (size_t)i * 768 * 3072, 768, 3072);
    gemm(h2, wbuf, ff2_b1 + oDF, nullptr, h1, 512, 3072, 768, 2, 1.f);
    cvtw(ff2_w2 + (size_t)i * 3072 * 768, 3072, 768);
    gemm(h1, wbuf, ff2_b2 + oD, B, A, 512, 768, 3072, 0, 0.5f);
    // output LN
    layernorm_k<<<512, 256, 0, stream>>>(A, ln_out_g + oD, ln_out_b + oD, B);
    float* t = A; A = B; B = t;
  }

  transpose_out<<<(2 * 768 * 256 + 255) / 256, 256, 0, stream>>>(A,
                                                                 (float*)d_out);
}
